// GCN_79302276153963
// MI455X (gfx1250) — matrix operand order repaired and checked by reading, not compiled, not run on hardware
//
#include <hip/hip_runtime.h>
#include <math.h>

// ---------------------------------------------------------------------------
// GCN forward for MI455X (gfx1250).
// Dominant cost: two streaming passes over the 256MB f32 adjacency matrix,
// consumed as f16 WMMA A-operands (converted in registers, f32 accumulate).
// ---------------------------------------------------------------------------

typedef __attribute__((ext_vector_type(16))) _Float16 v16h;
typedef __attribute__((ext_vector_type(8)))  float    v8f;

static constexpr int NN     = 8192;   // nodes
static constexpr int NFEAT  = 128;
static constexpr int NHID   = 128;
static constexpr int NCLASS = 40;
static constexpr int NPAD   = 64;     // padded class dim for WMMA pass 2

// ---------------- small f32 GEMMs (negligible cost, keep full precision) ---

// h = x @ W_emb + b_emb      (one thread per output element)
__global__ void k_emb(const float* __restrict__ x, const float* __restrict__ W,
                      const float* __restrict__ b, float* __restrict__ h) {
  int idx = blockIdx.x * blockDim.x + threadIdx.x;   // 8192*128 threads
  int row = idx >> 7, col = idx & 127;
  const float* xr = x + (size_t)row * NFEAT;
  float acc = b[col];
#pragma unroll 8
  for (int k = 0; k < NFEAT; ++k) acc += xr[k] * W[k * NHID + col];
  h[idx] = acc;
}

// s_src = h@We[:128];  BT[c][j]     = (h@We[128:])[j][c]  (f16, transposed)
//                      BT[128+c][j] = (h@Wgc1)[j][c]      (f16, transposed)
__global__ void k_branch(const float* __restrict__ h, const float* __restrict__ We,
                         const float* __restrict__ Wgc1,
                         float* __restrict__ s_src, _Float16* __restrict__ BT) {
  int idx = blockIdx.x * blockDim.x + threadIdx.x;   // 8192*128 threads
  int row = idx >> 7, col = idx & 127;
  const float* hr = h + (size_t)row * NHID;
  float a_src = 0.f, a_dst = 0.f, a_g1 = 0.f;
#pragma unroll 4
  for (int k = 0; k < NHID; ++k) {
    float hv = hr[k];
    a_src += hv * We[k * NHID + col];
    a_dst += hv * We[(NHID + k) * NHID + col];
    a_g1  += hv * Wgc1[k * NHID + col];
  }
  s_src[idx] = a_src;
  BT[(size_t)col * NN + row]           = (_Float16)a_dst;
  BT[(size_t)(128 + col) * NN + row]   = (_Float16)a_g1;
}

// ---------------- WMMA pass 1: P = adj @ [s_dst | g1], deg = rowsum(adj) ---
// Grid: 128 blocks x 256 threads (8 waves). Block tile: 64 rows x 256 cols.
// Wave tile: 16 rows x 128 cols = 8 WMMA accum tiles. adj read exactly once.
__global__ void __launch_bounds__(256) k_spmm1(const float* __restrict__ adj,
                                               const _Float16* __restrict__ BT,
                                               float* __restrict__ P,
                                               float* __restrict__ deg) {
  const int lid   = threadIdx.x & 31;
  const int wave  = threadIdx.x >> 5;   // 0..7
  const int mwave = wave >> 1;          // 0..3  (row strip)
  const int nwave = wave & 1;           // 0..1  (128-col half)
  const int g = lid >> 4;               // half-wave
  const int m = lid & 15;               // A row within tile / B col within tile
  const int R = blockIdx.x * 64 + mwave * 16;
  const int C = nwave * 128;

  const float*    arow  = adj + (size_t)(R + m) * NN + 8 * g;   // A chunks at +0,+16
  const _Float16* bbase = BT + (size_t)(C + m) * NN + 8 * g;    // B chunks at +0,+16

  v8f acc[8];
#pragma unroll
  for (int t = 0; t < 8; ++t) acc[t] = v8f{};
  float dsum = 0.f;

  for (int k0 = 0; k0 < NN; k0 += 32) {
    // ---- A fragment: 16 adj f32 -> f16 (rows of adj), per-lane row sum ----
    float4 f0 = *(const float4*)(arow + k0);
    float4 f1 = *(const float4*)(arow + k0 + 4);
    float4 f2 = *(const float4*)(arow + k0 + 16);
    float4 f3 = *(const float4*)(arow + k0 + 20);
    __builtin_prefetch(arow + k0 + 256, 0, 1);   // global_prefetch ahead
    dsum += (f0.x + f0.y + f0.z + f0.w) + (f1.x + f1.y + f1.z + f1.w)
          + (f2.x + f2.y + f2.z + f2.w) + (f3.x + f3.y + f3.z + f3.w);
    v16h a;
    a[0]=(_Float16)f0.x; a[1]=(_Float16)f0.y; a[2]=(_Float16)f0.z; a[3]=(_Float16)f0.w;
    a[4]=(_Float16)f1.x; a[5]=(_Float16)f1.y; a[6]=(_Float16)f1.z; a[7]=(_Float16)f1.w;
    a[8]=(_Float16)f2.x; a[9]=(_Float16)f2.y; a[10]=(_Float16)f2.z; a[11]=(_Float16)f2.w;
    a[12]=(_Float16)f3.x; a[13]=(_Float16)f3.y; a[14]=(_Float16)f3.z; a[15]=(_Float16)f3.w;
    // ---- 8 N-tiles: B fragment = 2x b128, then WMMA f32 <- f16*f16 -------
#pragma unroll
    for (int nt = 0; nt < 8; ++nt) {
      const _Float16* bp = bbase + (size_t)nt * 16 * NN + k0;
      v16h b;
      ((uint4*)&b)[0] = *(const uint4*)(bp);
      ((uint4*)&b)[1] = *(const uint4*)(bp + 16);
      acc[nt] = __builtin_amdgcn_wmma_f32_16x16x32_f16(
          false, a, false, b, (short)0, acc[nt], false, false);
    }
  }

  // deg: lane m holds K chunks {8g..}+{16+8g..}; combine g-halves, lanes 0..15 write
  if (nwave == 0) {
    dsum += __shfl_xor(dsum, 16, 32);
    if (lid < 16) deg[R + m] = dsum;
  }

  // store accumulators: lane 16g+n holds D[8g+r][n] of each 16x16 tile
#pragma unroll
  for (int nt = 0; nt < 8; ++nt)
#pragma unroll
    for (int r = 0; r < 8; ++r)
      P[(size_t)(R + 8 * g + r) * 256 + C + nt * 16 + m] = acc[nt][r];
}

// ---------------- epilogue 1: x_e out, h1 = relu(...) ---------------------
__global__ void k_epi1(const float* __restrict__ P, const float* __restrict__ deg,
                       const float* __restrict__ s_src, const float* __restrict__ b_e,
                       const float* __restrict__ b_gc1,
                       float* __restrict__ xe_out, float* __restrict__ h1) {
  int idx = blockIdx.x * blockDim.x + threadIdx.x;   // 8192*128
  int row = idx >> 7, col = idx & 127;
  xe_out[idx] = deg[row] * s_src[idx] + P[(size_t)row * 256 + col] + b_e[col];
  float v = P[(size_t)row * 256 + 128 + col] + b_gc1[col];
  h1[idx] = v > 0.f ? v : 0.f;
}

// zT[c][j] = (h1 @ W_gc2)[j][c] for c<40, 0 otherwise (f16, transposed, padded)
__global__ void k_zt(const float* __restrict__ h1, const float* __restrict__ Wgc2,
                     _Float16* __restrict__ zT) {
  int idx = blockIdx.x * blockDim.x + threadIdx.x;   // 8192*64
  int row = idx >> 6, col = idx & 63;
  float acc = 0.f;
  if (col < NCLASS) {
    const float* hr = h1 + (size_t)row * NHID;
#pragma unroll 4
    for (int k = 0; k < NHID; ++k) acc += hr[k] * Wgc2[k * NCLASS + col];
  }
  zT[(size_t)col * NN + row] = (_Float16)acc;
}

// ---------------- WMMA pass 2: logits = adj @ z ---------------------------
// Grid: 64 blocks x 256 threads. Block tile: 128 rows x 64 cols; wave: 16x64.
__global__ void __launch_bounds__(256) k_spmm2(const float* __restrict__ adj,
                                               const _Float16* __restrict__ zT,
                                               float* __restrict__ logits) {
  const int lid  = threadIdx.x & 31;
  const int wave = threadIdx.x >> 5;            // 0..7 -> row strip
  const int g = lid >> 4, m = lid & 15;
  const int R = blockIdx.x * 128 + wave * 16;

  const float*    arow  = adj + (size_t)(R + m) * NN + 8 * g;
  const _Float16* bbase = zT + (size_t)m * NN + 8 * g;

  v8f acc[4];
#pragma unroll
  for (int t = 0; t < 4; ++t) acc[t] = v8f{};

  for (int k0 = 0; k0 < NN; k0 += 32) {
    float4 f0 = *(const float4*)(arow + k0);
    float4 f1 = *(const float4*)(arow + k0 + 4);
    float4 f2 = *(const float4*)(arow + k0 + 16);
    float4 f3 = *(const float4*)(arow + k0 + 20);
    __builtin_prefetch(arow + k0 + 256, 0, 1);
    v16h a;
    a[0]=(_Float16)f0.x; a[1]=(_Float16)f0.y; a[2]=(_Float16)f0.z; a[3]=(_Float16)f0.w;
    a[4]=(_Float16)f1.x; a[5]=(_Float16)f1.y; a[6]=(_Float16)f1.z; a[7]=(_Float16)f1.w;
    a[8]=(_Float16)f2.x; a[9]=(_Float16)f2.y; a[10]=(_Float16)f2.z; a[11]=(_Float16)f2.w;
    a[12]=(_Float16)f3.x; a[13]=(_Float16)f3.y; a[14]=(_Float16)f3.z; a[15]=(_Float16)f3.w;
#pragma unroll
    for (int nt = 0; nt < 4; ++nt) {
      const _Float16* bp = bbase + (size_t)nt * 16 * NN + k0;
      v16h b;
      ((uint4*)&b)[0] = *(const uint4*)(bp);
      ((uint4*)&b)[1] = *(const uint4*)(bp + 16);
      acc[nt] = __builtin_amdgcn_wmma_f32_16x16x32_f16(
          false, a, false, b, (short)0, acc[nt], false, false);
    }
  }
#pragma unroll
  for (int nt = 0; nt < 4; ++nt)
#pragma unroll
    for (int r = 0; r < 8; ++r)
      logits[(size_t)(R + 8 * g + r) * NPAD + nt * 16 + m] = acc[nt][r];
}

// ---------------- log_softmax over 40 classes, one wave per row -----------
__global__ void k_lsm(const float* __restrict__ logits, const float* __restrict__ b_gc2,
                      float* __restrict__ out) {
  int wave = threadIdx.x >> 5, lane = threadIdx.x & 31;
  int row = blockIdx.x * 8 + wave;
  const float NEG = -3.0e38f;
  int c1 = lane + 32;
  float v0 = (lane < NCLASS) ? logits[(size_t)row * NPAD + lane] + b_gc2[lane] : NEG;
  float v1 = (c1 < NCLASS)   ? logits[(size_t)row * NPAD + c1] + b_gc2[c1]     : NEG;
  float mx = fmaxf(v0, v1);
#pragma unroll
  for (int off = 16; off >= 1; off >>= 1) mx = fmaxf(mx, __shfl_xor(mx, off, 32));
  float s = ((lane < NCLASS) ? __expf(v0 - mx) : 0.f) + ((c1 < NCLASS) ? __expf(v1 - mx) : 0.f);
#pragma unroll
  for (int off = 16; off >= 1; off >>= 1) s += __shfl_xor(s, off, 32);
  float lse = __logf(s);
  if (lane < NCLASS) out[(size_t)row * NCLASS + lane] = v0 - mx - lse;
  if (c1 < NCLASS)   out[(size_t)row * NCLASS + c1]   = v1 - mx - lse;
}

// ---------------------------------------------------------------------------
extern "C" void kernel_launch(void* const* d_in, const int* in_sizes, int n_in,
                              void* d_out, int out_size, void* d_ws, size_t ws_size,
                              hipStream_t stream) {
  const float* x     = (const float*)d_in[0];
  const float* adj   = (const float*)d_in[1];
  const float* W_emb = (const float*)d_in[2];
  const float* b_emb = (const float*)d_in[3];
  const float* W_gc1 = (const float*)d_in[4];
  const float* b_gc1 = (const float*)d_in[5];
  const float* W_e   = (const float*)d_in[6];
  const float* b_e   = (const float*)d_in[7];
  const float* W_gc2 = (const float*)d_in[8];
  const float* b_gc2 = (const float*)d_in[9];
  (void)in_sizes; (void)n_in; (void)out_size; (void)ws_size;

  // outputs: (log_softmax [8192,40], x_e [8192,128]) concatenated
  float* out_lsm = (float*)d_out;
  float* out_xe  = out_lsm + (size_t)NN * NCLASS;

  // workspace carve-up (~28.3 MB)
  float* ws      = (float*)d_ws;
  float* h       = ws;                          // 8192*128
  float* s_src   = h + (size_t)NN * NHID;       // 8192*128
  float* h1      = s_src + (size_t)NN * NHID;   // 8192*128
  float* P       = h1 + (size_t)NN * NHID;      // 8192*256
  float* logits  = P + (size_t)NN * 256;        // 8192*64
  float* deg     = logits + (size_t)NN * NPAD;  // 8192
  _Float16* BT   = (_Float16*)(deg + NN);       // 256*8192 f16
  _Float16* zT   = BT + (size_t)256 * NN;       // 64*8192 f16

  dim3 blk(256);
  k_emb   <<<dim3((NN * NHID) / 256), blk, 0, stream>>>(x, W_emb, b_emb, h);
  k_branch<<<dim3((NN * NHID) / 256), blk, 0, stream>>>(h, W_e, W_gc1, s_src, BT);
  k_spmm1 <<<dim3(NN / 64),           blk, 0, stream>>>(adj, BT, P, deg);
  k_epi1  <<<dim3((NN * NHID) / 256), blk, 0, stream>>>(P, deg, s_src, b_e, b_gc1,
                                                        out_xe, h1);
  k_zt    <<<dim3((NN * NPAD) / 256), blk, 0, stream>>>(h1, W_gc2, zT);
  k_spmm2 <<<dim3(NN / 128),          blk, 0, stream>>>(adj, zT, logits);
  k_lsm   <<<dim3(NN / 8),            blk, 0, stream>>>(logits, b_gc2, out_lsm);
}
